// Attend_1726576856082
// MI455X (gfx1250) — matrix operand order repaired and checked by reading, not compiled, not run on hardware
//
#include <hip/hip_runtime.h>
#include <hip/hip_bf16.h>

typedef __attribute__((ext_vector_type(16))) _Float16 v16h;
typedef __attribute__((ext_vector_type(8)))  _Float16 v8h;
typedef __attribute__((ext_vector_type(8)))  float    v8f;
typedef __attribute__((ext_vector_type(4)))  int      v4i;
typedef __attribute__((address_space(1))) v4i* gptr4;   // global int4*
typedef __attribute__((address_space(3))) v4i* lptr4;   // LDS int4*

namespace {
constexpr int Bsz = 4, Hn = 8, Nn = 1024, Dd = 64;
constexpr int BQ = 64;   // query rows per block
constexpr int WQ = 16;   // query rows per wave (one WMMA M tile)
constexpr int TC = 32;   // keys per chunk (one WMMA K=32 for P*V)
}

#if __has_builtin(__builtin_amdgcn_global_load_async_to_lds_b128)
#define HAVE_ASYNC 1
#else
#define HAVE_ASYNC 0
#endif
#if __has_builtin(__builtin_amdgcn_permlane16)
#define HAVE_PERMLANE16 1
#else
#define HAVE_PERMLANE16 0
#endif

#if HAVE_ASYNC
#define ASYNC_COPY16(g, l) __builtin_amdgcn_global_load_async_to_lds_b128( \
    (gptr4)(void*)(g), (lptr4)(void*)(l), 0, 0)
#if __has_builtin(__builtin_amdgcn_s_wait_asynccnt)
#define ASYNC_WAIT(n) __builtin_amdgcn_s_wait_asynccnt(n)
#else
#define ASYNC_WAIT(n) asm volatile("s_wait_asynccnt %0" :: "i"(n) : "memory")
#endif
#else
#define ASYNC_COPY16(g, l) (*(v8h*)(l) = *(const v8h*)(g))
#define ASYNC_WAIT(n)
#endif

__device__ __forceinline__ v16h cat16(v8h lo, v8h hi) {
  v16h r;
#pragma unroll
  for (int i = 0; i < 8; ++i) { r[i] = lo[i]; r[i + 8] = hi[i]; }
  return r;
}

#if HAVE_PERMLANE16
__device__ __forceinline__ float permf16(float x, unsigned lo, unsigned hi) {
  unsigned u = __float_as_uint(x);
  u = __builtin_amdgcn_permlane16(u, u, lo, hi, false, false);
  return __uint_as_float(u);
}
#endif

// max-reduce across the 16-lane row group (lanes 0-15 / 16-31 independently)
__device__ __forceinline__ float rowmax16(float t) {
#if HAVE_PERMLANE16
  t = fmaxf(t, permf16(t, 0x67452301u, 0xEFCDAB89u)); // xor 1
  t = fmaxf(t, permf16(t, 0x54761032u, 0xDCFE98BAu)); // xor 2
  t = fmaxf(t, permf16(t, 0x32107654u, 0xBA98FEDCu)); // xor 4
  t = fmaxf(t, permf16(t, 0xFEDCBA98u, 0x76543210u)); // xor 8
#else
#pragma unroll
  for (int sh = 8; sh >= 1; sh >>= 1) t = fmaxf(t, __shfl_xor(t, sh, 32));
#endif
  return t;
}

// one-time K/V pre-pass: f32 -> f16, V additionally transposed to [B][D][N]
__global__ __launch_bounds__(256) void prep_kv_kernel(
    const float* __restrict__ k, const float* __restrict__ v,
    void* __restrict__ khv, void* __restrict__ vtv)
{
  _Float16* kh = (_Float16*)khv;
  _Float16* vt = (_Float16*)vtv;
  const size_t idx = (size_t)blockIdx.x * 256 + threadIdx.x;  // over B*N*D
  const int d = (int)(idx % Dd);
  const int j = (int)((idx / Dd) % Nn);
  const int b = (int)(idx / ((size_t)Dd * Nn));
  kh[idx] = (_Float16)k[idx];
  vt[((size_t)b * Dd + d) * Nn + j] = (_Float16)v[idx];
}

template <bool PRE>
__global__ __launch_bounds__(128) void fa_fwd_kernel(
    const float* __restrict__ q,
    const void* __restrict__ kv0,   // PRE ? kh f16 [B][N][D] : k f32 [B][N][D]
    const void* __restrict__ kv1,   // PRE ? vt f16 [B][D][N] : v f32 [B][N][D]
    const unsigned char* __restrict__ mask,
    const float* __restrict__ bias,
    float* __restrict__ out)
{
  // double-buffered K chunk: [j][d] f16 (stride 72 halves = 144B, 16B aligned)
  __shared__ __align__(16) _Float16 Ksh[2][TC][72];
  // double-buffered V chunk transposed: [d][j] f16 (stride 40 halves = 80B)
  __shared__ __align__(16) _Float16 Vsh[2][Dd][40];
  // per-wave P tile: [m][j] f16
  __shared__ __align__(16) _Float16 Psh[4][WQ][40];

  const int tile  = blockIdx.x;
  const int bh    = blockIdx.y;
  const int b     = bh / Hn;
  const int h     = bh % Hn;
  const int tid   = threadIdx.x;
  const int wave  = tid >> 5;
  const int lane  = tid & 31;
  const int lhalf = lane >> 4;
  const int l16   = lane & 15;
  const int m0    = tile * BQ;
  const int m0w   = m0 + wave * WQ;

  const float scale = 0.125f;               // 1/sqrt(64)
  const float NEG   = -3.402823466e38f;

  // per-thread staging geometry (constant across chunks)
  const int sj   = tid >> 3;                // K: row 0..15 (+16 with u)
  const int sseg = (tid & 7) * 8;           // K: halves 0..56
  const int sd   = tid >> 2;                // V: d 0..31 (+32 with u)
  const int sjs  = (tid & 3) * 8;           // V: halves 0,8,16,24
  const _Float16* khb = (const _Float16*)kv0 + (size_t)b * Nn * Dd;
  const _Float16* vtb = (const _Float16*)kv1 + (size_t)b * Dd * Nn;

  // issue one chunk's K/V transfers (4 x B128 per thread = 4 async instr/wave)
  auto issue_chunk = [&](int jbase, int buf) {
#pragma unroll
    for (int u = 0; u < 2; ++u) {
      const int j = sj + u * 16;            // 0..31
      const int d = sd + u * 32;            // 0..63
      ASYNC_COPY16(khb + ((size_t)(jbase + j) * Dd + sseg), &Ksh[buf][j][sseg]);
      ASYNC_COPY16(vtb + ((size_t)d * Nn + jbase + sjs),    &Vsh[buf][d][sjs]);
    }
  };

  // ---- Q tile in WMMA A-operand layout, pre-scaled, f32->f16 ----
  v16h qa[2];
  {
    const int m    = m0w + l16;
    const int base = lhalf ? 8 : 0;
    const float* qrow = q + (((size_t)(b * Hn + h) * Nn + m) * Dd);
#pragma unroll
    for (int c = 0; c < 2; ++c) {
#pragma unroll
      for (int e = 0; e < 8; ++e) {
        qa[c][e]     = (_Float16)(qrow[32 * c + base + e] * scale);
        qa[c][e + 8] = (_Float16)(qrow[32 * c + 16 + base + e] * scale);
      }
    }
  }

  // per-row bias pointers (i fixed per slot r across all chunks)
  const float* biasrow[8];
#pragma unroll
  for (int r = 0; r < 8; ++r) {
    const int i = m0w + r + 8 * lhalf;
    biasrow[r] = bias + ((size_t)(b * Hn + h) * Nn + i) * (size_t)Nn;
  }

  // all-ones B operand: WMMA row-sum trick (result replicated over columns,
  // lands in exactly the (slot r, lane-half) layout of lrow)
  v16h ones;
#pragma unroll
  for (int i = 0; i < 16; ++i) ones[i] = (_Float16)1.0f;

  v8f o[4] = {};
  float mrow[8], lrow[8];
#pragma unroll
  for (int r = 0; r < 8; ++r) { mrow[r] = NEG; lrow[r] = 0.f; }

  const int nchunks = (m0 + BQ) / TC;

  if (PRE) issue_chunk(0, 0);                 // pipeline prologue

  for (int ch = 0; ch < nchunks; ++ch) {
    const int jbase = ch * TC;
    const int buf   = ch & 1;

    if (PRE) {
      // prefetch next chunk into the other buffer, then retire current chunk
      // (ASYNCcnt completes in order: <=4 leaves only the prefetch in flight)
      if (ch + 1 < nchunks) { issue_chunk(jbase + TC, buf ^ 1); ASYNC_WAIT(4); }
      else                  { ASYNC_WAIT(0); }
    } else {
      const float* kf = (const float*)kv0;
      const float* vf = (const float*)kv1;
      const int j    = tid >> 2;
      const int dseg = (tid & 3) * 16;
      const float* krow = kf + ((size_t)b * Nn + (jbase + j)) * Dd + dseg;
      const float* vrow = vf + ((size_t)b * Nn + (jbase + j)) * Dd + dseg;
#pragma unroll
      for (int e = 0; e < 16; ++e) {
        Ksh[buf][j][dseg + e] = (_Float16)krow[e];
        Vsh[buf][dseg + e][j] = (_Float16)vrow[e];
      }
    }
    __syncthreads();

    if (jbase <= m0w + WQ - 1) {   // wave-level causal skip
      // ---- issue streaming bias loads early (hide behind S WMMAs) ----
      const int j0 = jbase + l16;
      const int j1 = j0 + 16;
      const bool km0 = mask[(size_t)b * Nn + j0] != 0;
      const bool km1 = mask[(size_t)b * Nn + j1] != 0;
      float bv0[8], bv1[8];
#pragma unroll
      for (int r = 0; r < 8; ++r) {
        bv0[r] = __builtin_nontemporal_load(biasrow[r] + j0);
        bv1[r] = __builtin_nontemporal_load(biasrow[r] + j1);
      }

      // ---- S = Q K^T : two 16x16 j-tiles, K-dim 64 = 2 x 32 ----
      const int koff = lhalf * 8;
      v8f c0 = {}, c1 = {};
#pragma unroll
      for (int c = 0; c < 2; ++c) {
        v16h bk0 = cat16(*(const v8h*)&Ksh[buf][l16     ][32 * c + koff],
                         *(const v8h*)&Ksh[buf][l16     ][32 * c + koff + 16]);
        v16h bk1 = cat16(*(const v8h*)&Ksh[buf][16 + l16][32 * c + koff],
                         *(const v8h*)&Ksh[buf][16 + l16][32 * c + koff + 16]);
        c0 = __builtin_amdgcn_wmma_f32_16x16x32_f16(false, qa[c], false, bk0,
                                                    (short)0, c0, false, false);
        c1 = __builtin_amdgcn_wmma_f32_16x16x32_f16(false, qa[c], false, bk1,
                                                    (short)0, c1, false, false);
      }

      // ---- masks + online softmax (max via permlane16, sum via WMMA) ----
#pragma unroll
      for (int r = 0; r < 8; ++r) {
        const int i = m0w + r + 8 * lhalf;
        const float s0 = (km0 && j0 <= i) ? (c0[r] + bv0[r]) : NEG;
        const float s1 = (km1 && j1 <= i) ? (c1[r] + bv1[r]) : NEG;
        const float mnew = fmaxf(mrow[r], rowmax16(fmaxf(s0, s1)));
        const float p0 = __expf(s0 - mnew);
        const float p1 = __expf(s1 - mnew);
        const float alpha = __expf(mrow[r] - mnew);
        mrow[r] = mnew;
        lrow[r] *= alpha;
#pragma unroll
        for (int t4 = 0; t4 < 4; ++t4) o[t4][r] *= alpha;
        const int m = r + 8 * lhalf;
        Psh[wave][m][l16]      = (_Float16)p0;
        Psh[wave][m][16 + l16] = (_Float16)p1;
      }

      // ---- reload P as A-operand; row sums + O += P V ----
      {
        const int base = lhalf ? 8 : 0;
        const v16h pa = cat16(*(const v8h*)&Psh[wave][l16][base],
                              *(const v8h*)&Psh[wave][l16][16 + base]);
        v8f zero = {};
        v8f lsum = __builtin_amdgcn_wmma_f32_16x16x32_f16(false, pa, false, ones,
                                                          (short)0, zero, false, false);
#pragma unroll
        for (int r = 0; r < 8; ++r) lrow[r] += lsum[r];
#pragma unroll
        for (int t4 = 0; t4 < 4; ++t4) {
          const v16h bv = cat16(*(const v8h*)&Vsh[buf][16 * t4 + l16][8 * lhalf],
                                *(const v8h*)&Vsh[buf][16 * t4 + l16][16 + 8 * lhalf]);
          o[t4] = __builtin_amdgcn_wmma_f32_16x16x32_f16(false, pa, false, bv,
                                                         (short)0, o[t4], false, false);
        }
      }
    }
    __syncthreads();
  }

  // ---- normalize and store ----
#pragma unroll
  for (int r = 0; r < 8; ++r) {
    const int i = m0w + r + 8 * lhalf;
    const float inv = 1.0f / lrow[r];
    float* orow = out + (((size_t)(b * Hn + h) * Nn + i) * Dd);
#pragma unroll
    for (int t4 = 0; t4 < 4; ++t4)
      __builtin_nontemporal_store(o[t4][r] * inv, &orow[16 * t4 + l16]);
  }
}

extern "C" void kernel_launch(void* const* d_in, const int* in_sizes, int n_in,
                              void* d_out, int out_size, void* d_ws, size_t ws_size,
                              hipStream_t stream) {
  const float* q = (const float*)d_in[0];
  const float* k = (const float*)d_in[1];
  const float* v = (const float*)d_in[2];
  const unsigned char* mask = (const unsigned char*)d_in[3];  // jax bool -> 1 byte
  const float* bias = (const float*)d_in[4];
  float* out = (float*)d_out;
  (void)in_sizes; (void)n_in; (void)out_size;

  dim3 grid(Nn / BQ, Bsz * Hn);
  const size_t half_bytes = (size_t)Bsz * Nn * Dd * 2;  // 512 KB per array
  if (ws_size >= 2 * half_bytes) {
    void* kh = d_ws;
    void* vt = (void*)((char*)d_ws + half_bytes);
    prep_kv_kernel<<<(Bsz * Nn * Dd) / 256, 256, 0, stream>>>(k, v, kh, vt);
    fa_fwd_kernel<true><<<grid, 128, 0, stream>>>(q, kh, vt, mask, bias, out);
  } else {
    fa_fwd_kernel<false><<<grid, 128, 0, stream>>>(q, k, v, mask, bias, out);
  }
}
